// SlidingWindowCrossAttention_90426241450524
// MI455X (gfx1250) — matrix operand order repaired and checked by reading, not compiled, not run on hardware
//
#include <hip/hip_runtime.h>
#include <hip/hip_bf16.h>

typedef __attribute__((ext_vector_type(16))) _Float16 v16h;
typedef __attribute__((ext_vector_type(8)))  _Float16 v8h;
typedef __attribute__((ext_vector_type(4)))  _Float16 v4h;
typedef __attribute__((ext_vector_type(8)))  float    v8f;

#define S_LEN   2048
#define DMODEL  1024
#define NHEADS  16
#define HDIM    64
#define SPAD    2176   // S_LEN + 128 (64 zero rows front, 64 back)
#define PADL    64

static __device__ __forceinline__ v16h hcat(v8h lo, v8h hi) {
  return __builtin_shufflevector(lo, hi, 0,1,2,3,4,5,6,7,8,9,10,11,12,13,14,15);
}
static __device__ __forceinline__ v8h ld8(const _Float16* p) { return *(const v8h*)p; }

// ---------------------------------------------------------------------------
// fp32 -> fp16 conversion (vectorized: 4 elems / thread)
// ---------------------------------------------------------------------------
__global__ void cvt_f32_f16(const float* __restrict__ in, _Float16* __restrict__ out, int n4) {
  int i = blockIdx.x * blockDim.x + threadIdx.x;
  if (i >= n4) return;
  float4 v = ((const float4*)in)[i];
  v4h o;
  o.x = (_Float16)v.x; o.y = (_Float16)v.y; o.z = (_Float16)v.z; o.w = (_Float16)v.w;
  ((v4h*)out)[i] = o;
}

// zero fp16 buffer (16B granules)
__global__ void zero16(int4* p, int n16) {
  int i = blockIdx.x * blockDim.x + threadIdx.x;
  if (i >= n16) return;
  p[i] = make_int4(0, 0, 0, 0);
}

// ---------------------------------------------------------------------------
// GEMM: out[M,N] (f32) = X[M,K](f16) @ W[N,K]^T (f16) + bias
// wave = 32x64 output strip: 2 A-fragments x 4 B-fragments -> 8 accumulators.
// K-step 32 via v_wmma_f32_16x16x32_f16. 12 b128 loads : 8 WMMAs per step.
// ---------------------------------------------------------------------------
__global__ __launch_bounds__(256) void gemm_bias_wmma(
    const _Float16* __restrict__ X, const _Float16* __restrict__ Wt,
    const float* __restrict__ bias, float* __restrict__ out,
    int M, int N, int K)
{
  const int lane = threadIdx.x & 31;
  const int wid  = blockIdx.x * 8 + (threadIdx.x >> 5);
  const int ngrp = N >> 6;                 // groups of 64 cols
  const int mt   = wid / ngrp;             // 32-row tiles
  const int ng   = wid % ngrp;
  if (mt * 32 >= M) return;

  const int l15  = lane & 15;
  const int half = lane >> 4;
  const int akb  = half * 16;              // A K-base within 32
  const int bkb  = half * 16;              // B K-base within 32

  const _Float16* xr0 = X + (size_t)(mt * 32 + l15) * K;
  const _Float16* xr1 = xr0 + (size_t)16 * K;
  const int c0 = ng * 64 + l15;
  const _Float16* w0 = Wt + (size_t)(c0 +  0) * K;
  const _Float16* w1 = Wt + (size_t)(c0 + 16) * K;
  const _Float16* w2 = Wt + (size_t)(c0 + 32) * K;
  const _Float16* w3 = Wt + (size_t)(c0 + 48) * K;

  v8f p0 = {}, p1 = {}, p2 = {}, p3 = {};   // rows [mt*32 .. +15]
  v8f q0 = {}, q1 = {}, q2 = {}, q3 = {};   // rows [mt*32+16 .. +31]
  for (int k0 = 0; k0 < K; k0 += 32) {
    v16h aLo = hcat(ld8(xr0 + k0 + akb), ld8(xr0 + k0 + akb + 8));
    v16h aHi = hcat(ld8(xr1 + k0 + akb), ld8(xr1 + k0 + akb + 8));
    v16h b0  = hcat(ld8(w0 + k0 + bkb),  ld8(w0 + k0 + bkb + 8));
    v16h b1  = hcat(ld8(w1 + k0 + bkb),  ld8(w1 + k0 + bkb + 8));
    v16h b2  = hcat(ld8(w2 + k0 + bkb),  ld8(w2 + k0 + bkb + 8));
    v16h b3  = hcat(ld8(w3 + k0 + bkb),  ld8(w3 + k0 + bkb + 8));
    p0 = __builtin_amdgcn_wmma_f32_16x16x32_f16(false, aLo, false, b0, (short)0, p0, false, false);
    q0 = __builtin_amdgcn_wmma_f32_16x16x32_f16(false, aHi, false, b0, (short)0, q0, false, false);
    p1 = __builtin_amdgcn_wmma_f32_16x16x32_f16(false, aLo, false, b1, (short)0, p1, false, false);
    q1 = __builtin_amdgcn_wmma_f32_16x16x32_f16(false, aHi, false, b1, (short)0, q1, false, false);
    p2 = __builtin_amdgcn_wmma_f32_16x16x32_f16(false, aLo, false, b2, (short)0, p2, false, false);
    q2 = __builtin_amdgcn_wmma_f32_16x16x32_f16(false, aHi, false, b2, (short)0, q2, false, false);
    p3 = __builtin_amdgcn_wmma_f32_16x16x32_f16(false, aLo, false, b3, (short)0, p3, false, false);
    q3 = __builtin_amdgcn_wmma_f32_16x16x32_f16(false, aHi, false, b3, (short)0, q3, false, false);
  }

  const float b0s = bias[c0], b1s = bias[c0 + 16], b2s = bias[c0 + 32], b3s = bias[c0 + 48];
#pragma unroll
  for (int r = 0; r < 8; ++r) {
    float* op = out + (size_t)(mt * 32 + r + 8 * half) * N + ng * 64 + l15;
    op[0]  = p0[r] + b0s;
    op[16] = p1[r] + b1s;
    op[32] = p2[r] + b2s;
    op[48] = p3[r] + b3s;
    float* oq = op + (size_t)16 * N;
    oq[0]  = q0[r] + b0s;
    oq[16] = q1[r] + b1s;
    oq[32] = q2[r] + b2s;
    oq[48] = q3[r] + b3s;
  }
}

// ---------------------------------------------------------------------------
// RoPE + scale + pack:  qlin/klin/vlin [S][D] f32 ->
//   qh [H][S][64] f16 (rope * d^-1/2), kh [H][SPAD][64] f16 (rope, +64 pad),
//   vT [H][64][SPAD] f16 (transposed, +64 pad)
// ---------------------------------------------------------------------------
__global__ __launch_bounds__(512) void rope_pack(
    const float* __restrict__ ql, const float* __restrict__ kl,
    const float* __restrict__ vl,
    _Float16* __restrict__ qh, _Float16* __restrict__ kh, _Float16* __restrict__ vT)
{
  const int s = blockIdx.x;
  const int t = threadIdx.x;       // 0..511
  const int h = t >> 5;            // head 0..15
  const int i = t & 31;            // rotary pair 0..31

  const float freq = __powf(10000.f, -(float)i * (1.0f / 32.0f));
  float sn, cs;
  __sincosf((float)s * freq, &sn, &cs);

  const size_t base = (size_t)s * DMODEL + h * HDIM;
  const float q1 = ql[base + i], q2 = ql[base + 32 + i];
  const float k1 = kl[base + i], k2 = kl[base + 32 + i];
  const float rs = 0.125f;         // 64^-0.5

  _Float16* qr = qh + ((size_t)h * S_LEN + s) * HDIM;
  qr[i]      = (_Float16)((q1 * cs - q2 * sn) * rs);
  qr[32 + i] = (_Float16)((q1 * sn + q2 * cs) * rs);

  _Float16* kr = kh + ((size_t)h * SPAD + (s + PADL)) * HDIM;
  kr[i]      = (_Float16)(k1 * cs - k2 * sn);
  kr[32 + i] = (_Float16)(k1 * sn + k2 * cs);

  vT[((size_t)h * HDIM + i)      * SPAD + PADL + s] = (_Float16)vl[base + i];
  vT[((size_t)h * HDIM + 32 + i) * SPAD + PADL + s] = (_Float16)vl[base + 32 + i];
}

// ---------------------------------------------------------------------------
// Sliding-window attention. One wave = (head, 16-query tile).
// scores 16x80 (5 key tiles x 2 wmma) -> LDS -> softmax (2 lanes/row,
// combine via __shfl_xor) -> probs f16 LDS -> P@V (3 K-chunks x 4 N-tiles).
// ---------------------------------------------------------------------------
#define ATT_WAVES 4
__global__ __launch_bounds__(ATT_WAVES * 32) void attn_window_wmma(
    const _Float16* __restrict__ qh, const _Float16* __restrict__ kh,
    const _Float16* __restrict__ vT, _Float16* __restrict__ ctx)
{
  __shared__ __align__(16) float    sc[ATT_WAVES][16 * 96];
  __shared__ __align__(16) _Float16 pb[ATT_WAVES][16 * 96];

  const int lane = threadIdx.x & 31;
  const int wv   = threadIdx.x >> 5;
  const int wid  = blockIdx.x * ATT_WAVES + wv;
  const int h    = wid >> 7;               // 128 q-tiles per head
  const int q0   = (wid & 127) * 16;
  const int l15  = lane & 15;
  const int half = lane >> 4;
  const int akb  = half * 16;
  const int bkb  = half * 16;

  // Q fragments (16x64 tile, two K-steps of 32)
  const _Float16* qrow = qh + ((size_t)h * S_LEN + (q0 + l15)) * HDIM;
  v16h aq0 = hcat(ld8(qrow + akb),      ld8(qrow + akb + 8));
  v16h aq1 = hcat(ld8(qrow + 32 + akb), ld8(qrow + 32 + akb + 8));

  float* mysc = sc[wv];
#pragma unroll
  for (int kt = 0; kt < 5; ++kt) {
    // padded key row index: (q0-64 + kt*16 + n) + PADL = q0 + kt*16 + n
    const _Float16* krow = kh + ((size_t)h * SPAD + (q0 + kt * 16 + l15)) * HDIM;
    v16h b0 = hcat(ld8(krow + bkb),      ld8(krow + bkb + 8));
    v16h b1 = hcat(ld8(krow + 32 + bkb), ld8(krow + 32 + bkb + 8));
    v8f acc = {};
    acc = __builtin_amdgcn_wmma_f32_16x16x32_f16(false, aq0, false, b0, (short)0, acc, false, false);
    acc = __builtin_amdgcn_wmma_f32_16x16x32_f16(false, aq1, false, b1, (short)0, acc, false, false);
#pragma unroll
    for (int r = 0; r < 8; ++r)
      mysc[(r + 8 * half) * 96 + kt * 16 + l15] = acc[r];
  }
  __syncthreads();

  // softmax: rows 0..15, two lanes per row (each covers 48 of 96 window slots)
  {
    const int row   = l15;
    const int wbase = half * 48;
    int lo = row;                          // key >= q-64
    if (PADL - q0 > lo) lo = PADL - q0;    // key >= 0
    const int hi = row + 64;               // key <= q (and key < S always holds)
    const float* srow = mysc + row * 96;

    float mx = -1e30f;
    for (int w = wbase; w < wbase + 48; ++w)
      if (w >= lo && w <= hi) mx = fmaxf(mx, srow[w]);
    mx = fmaxf(mx, __shfl_xor(mx, 16, 32));

    float sum = 0.f;
    for (int w = wbase; w < wbase + 48; ++w)
      if (w >= lo && w <= hi) sum += __expf(srow[w] - mx);
    sum += __shfl_xor(sum, 16, 32);
    const float inv = 1.0f / sum;

    _Float16* prow = pb[wv] + row * 96;
    for (int w = wbase; w < wbase + 48; ++w) {
      float p = (w >= lo && w <= hi) ? __expf(srow[w] - mx) * inv : 0.0f;
      prow[w] = (_Float16)p;
    }
  }
  __syncthreads();

  // P @ V : probs 16x96 (zero padded past 80) x V-window 96x64
  v8f c0 = {}, c1 = {}, c2 = {}, c3 = {};
  const _Float16* myp = pb[wv];
#pragma unroll
  for (int ch = 0; ch < 3; ++ch) {
    const _Float16* prow = myp + l15 * 96 + ch * 32 + akb;
    v16h ap = hcat(ld8(prow), ld8(prow + 8));
#pragma unroll
    for (int nt = 0; nt < 4; ++nt) {
      // padded col = (q0-64 + w) + PADL = q0 + w ;  w = ch*32 + bkb + 0..15
      const _Float16* vrow = vT + ((size_t)h * HDIM + nt * 16 + l15) * SPAD
                                + q0 + ch * 32 + bkb;
      v16h b = hcat(ld8(vrow), ld8(vrow + 8));
      v8f& acc = (nt == 0) ? c0 : (nt == 1) ? c1 : (nt == 2) ? c2 : c3;
      acc = __builtin_amdgcn_wmma_f32_16x16x32_f16(false, ap, false, b, (short)0, acc, false, false);
    }
  }

  // store ctx (f16) in [S][D] layout for the output projection GEMM
#pragma unroll
  for (int r = 0; r < 8; ++r) {
    _Float16* op = ctx + (size_t)(q0 + r + 8 * half) * DMODEL + h * HDIM + l15;
    op[0]  = (_Float16)c0[r];
    op[16] = (_Float16)c1[r];
    op[32] = (_Float16)c2[r];
    op[48] = (_Float16)c3[r];
  }
}

// ---------------------------------------------------------------------------
// host-side launcher
// ---------------------------------------------------------------------------
extern "C" void kernel_launch(void* const* d_in, const int* in_sizes, int n_in,
                              void* d_out, int out_size, void* d_ws, size_t ws_size,
                              hipStream_t stream) {
  const float* q_in = (const float*)d_in[0];
  const float* k_in = (const float*)d_in[1];
  const float* v_in = (const float*)d_in[2];
  const float* q_w  = (const float*)d_in[3];
  const float* q_b  = (const float*)d_in[4];
  const float* k_w  = (const float*)d_in[5];
  const float* k_b  = (const float*)d_in[6];
  const float* v_w  = (const float*)d_in[7];
  const float* v_b  = (const float*)d_in[8];
  const float* o_w  = (const float*)d_in[9];
  const float* o_b  = (const float*)d_in[10];

  char* ws = (char*)d_ws;
  size_t off = 0;
  auto alloc = [&](size_t bytes) -> void* {
    off = (off + 255) & ~(size_t)255;
    void* p = ws + off;
    off += bytes;
    return p;
  };

  const size_t actN = (size_t)S_LEN * DMODEL;     // 2M
  const size_t wN   = (size_t)DMODEL * DMODEL;    // 1M

  _Float16* xq16 = (_Float16*)alloc(actN * 2);
  _Float16* xk16 = (_Float16*)alloc(actN * 2);
  _Float16* xv16 = (_Float16*)alloc(actN * 2);
  _Float16* qw16 = (_Float16*)alloc(wN * 2);
  _Float16* kw16 = (_Float16*)alloc(wN * 2);
  _Float16* vw16 = (_Float16*)alloc(wN * 2);
  _Float16* ow16 = (_Float16*)alloc(wN * 2);
  float*    qlin = (float*)alloc(actN * 4);
  float*    klin = (float*)alloc(actN * 4);
  float*    vlin = (float*)alloc(actN * 4);
  _Float16* qh   = (_Float16*)alloc((size_t)NHEADS * S_LEN * HDIM * 2);
  _Float16* kh   = (_Float16*)alloc((size_t)NHEADS * SPAD * HDIM * 2);
  _Float16* vT   = (_Float16*)alloc((size_t)NHEADS * HDIM * SPAD * 2);
  _Float16* c16  = (_Float16*)alloc(actN * 2);

  // 1) fp32 -> fp16 conversions
  {
    int n4 = (int)(actN / 4);
    int blk = (n4 + 255) / 256;
    cvt_f32_f16<<<blk, 256, 0, stream>>>(q_in, xq16, n4);
    cvt_f32_f16<<<blk, 256, 0, stream>>>(k_in, xk16, n4);
    cvt_f32_f16<<<blk, 256, 0, stream>>>(v_in, xv16, n4);
    int w4 = (int)(wN / 4);
    int wb = (w4 + 255) / 256;
    cvt_f32_f16<<<wb, 256, 0, stream>>>(q_w, qw16, w4);
    cvt_f32_f16<<<wb, 256, 0, stream>>>(k_w, kw16, w4);
    cvt_f32_f16<<<wb, 256, 0, stream>>>(v_w, vw16, w4);
    cvt_f32_f16<<<wb, 256, 0, stream>>>(o_w, ow16, w4);
  }

  // 2) zero the padded K / V^T buffers
  {
    int n16 = (int)(((size_t)NHEADS * SPAD * HDIM * 2) / 16);
    zero16<<<(n16 + 255) / 256, 256, 0, stream>>>((int4*)kh, n16);
    zero16<<<(n16 + 255) / 256, 256, 0, stream>>>((int4*)vT, n16);
  }

  // 3) Q / K / V projections (32x64 per wave: M/32 * N/64 waves)
  {
    int waves  = (S_LEN / 32) * (DMODEL / 64);  // 1024
    int blocks = waves / 8;                     // 128
    gemm_bias_wmma<<<blocks, 256, 0, stream>>>(xq16, qw16, q_b, qlin, S_LEN, DMODEL, DMODEL);
    gemm_bias_wmma<<<blocks, 256, 0, stream>>>(xk16, kw16, k_b, klin, S_LEN, DMODEL, DMODEL);
    gemm_bias_wmma<<<blocks, 256, 0, stream>>>(xv16, vw16, v_b, vlin, S_LEN, DMODEL, DMODEL);
  }

  // 4) RoPE + scale + head-major packing (V transposed)
  rope_pack<<<S_LEN, 512, 0, stream>>>(qlin, klin, vlin, qh, kh, vT);

  // 5) sliding-window attention
  {
    int waves  = NHEADS * (S_LEN / 16);         // 2048
    int blocks = waves / ATT_WAVES;             // 512
    attn_window_wmma<<<blocks, ATT_WAVES * 32, 0, stream>>>(qh, kh, vT, c16);
  }

  // 6) output projection -> d_out (f32)
  {
    int waves  = (S_LEN / 32) * (DMODEL / 64);
    int blocks = waves / 8;
    gemm_bias_wmma<<<blocks, 256, 0, stream>>>(c16, ow16, o_b, (float*)d_out, S_LEN, DMODEL, DMODEL);
  }
}
